// CrossAttentionBlock_56813827392085
// MI455X (gfx1250) — matrix operand order repaired and checked by reading, not compiled, not run on hardware
//
#include <hip/hip_runtime.h>
#include <hip/hip_bf16.h>
#include <math.h>

typedef __attribute__((ext_vector_type(16))) _Float16 v16h;
typedef __attribute__((ext_vector_type(8)))  _Float16 v8h;
typedef __attribute__((ext_vector_type(8)))  float    v8f;

#define BDIM 256

__device__ __forceinline__ v8f wmma_f16(v16h a, v16h b, v8f c) {
  return __builtin_amdgcn_wmma_f32_16x16x32_f16(false, a, false, b, (short)0, c,
                                                false, false);
}

// Fragment-major LDS: tile slot = ((tile*32 + lane) * 16) halves; each lane's
// 16 halves are contiguous 32 bytes -> 2x ds_load_b128 per fragment.
__device__ __forceinline__ v16h ld_frag(const _Float16* base, int tile, int lane) {
  return *(const v16h*)(base + ((size_t)(tile * 32 + lane)) * 16);
}

__device__ __forceinline__ float gelu_tanh(float v) {
  return 0.5f * v * (1.0f + tanhf(0.7978845608028654f * (v + 0.044715f * v * v * v)));
}

// ---------------------------------------------------------------------------
// Y[M,N] = epilogue( X[M,K] @ W[N,K]^T + bias )  mode 0: none, 1: gelu, 2: +res
// 128x64 tile per block, 8 waves; wave = 2 m-tiles x 2 n-tiles = 4 WMMA/kstep.
// Double-buffered LDS: one barrier per k-step; next step's global loads issue
// before the current step's WMMAs.
//
// A-frag map (16x32 f16): (m,k) -> lane m+16*((k>>3)&1), slot ((k>>4)<<3)+(k&7)
// B-frag map (32x16 f16): (k,n) -> lane n+16*((k>>3)&1), slot ((k>>4)<<3)+(k&7)
// ---------------------------------------------------------------------------
__global__ __launch_bounds__(BDIM) void gemm_kernel(
    const float* __restrict__ X, const float* __restrict__ W,
    const float* __restrict__ bias, const float* __restrict__ res,
    float* __restrict__ Y, int M, int N, int K, int mode) {
  __shared__ alignas(32) _Float16 Ash[2][8 * 32 * 16];
  __shared__ alignas(32) _Float16 Bsh[2][4 * 32 * 16];
  const int tid = threadIdx.x, lane = tid & 31, wave = tid >> 5;
  const int tm0 = (wave >> 1) * 2;
  const int tn0 = (wave & 1) * 2;
  const int bm = blockIdx.x * 128, bn = blockIdx.y * 64;
  // staging roles (constant across k-steps)
  const int ar0 = tid >> 2, akb0 = tid & 3;
  const int ar1 = (tid + BDIM) >> 2, akb1 = (tid + BDIM) & 3;
  const int bn_ = tid >> 2, bkb = tid & 3;
  const int adst0 = ((ar0 >> 4) * 32 + (ar0 & 15) + 16 * (akb0 & 1)) * 16 + (akb0 >> 1) * 8;
  const int adst1 = ((ar1 >> 4) * 32 + (ar1 & 15) + 16 * (akb1 & 1)) * 16 + (akb1 >> 1) * 8;
  const int bdst  = ((bn_ >> 4) * 32 + (bn_ & 15) + 16 * (bkb & 1)) * 16 + (bkb >> 1) * 8;

  auto stage_load = [&](int kb, v8h& ha0, v8h& ha1, v8h& hb) {
    const float* pa0 = X + (size_t)(bm + ar0) * K + kb + akb0 * 8;
    const float* pa1 = X + (size_t)(bm + ar1) * K + kb + akb1 * 8;
    const float* pb  = W + (size_t)(bn + bn_) * K + kb + bkb * 8;
#pragma unroll
    for (int e = 0; e < 8; ++e) {
      ha0[e] = (_Float16)pa0[e];
      ha1[e] = (_Float16)pa1[e];
      hb[e]  = (_Float16)pb[e];
    }
  };
  auto stage_store = [&](int buf, const v8h& ha0, const v8h& ha1, const v8h& hb) {
    *(v8h*)(Ash[buf] + adst0) = ha0;
    *(v8h*)(Ash[buf] + adst1) = ha1;
    *(v8h*)(Bsh[buf] + bdst)  = hb;
  };

  v8f acc[2][2] = {};
  const int nsteps = K >> 5;
  int cur = 0;
  {
    v8h ha0, ha1, hb;
    stage_load(0, ha0, ha1, hb);
    stage_store(0, ha0, ha1, hb);
  }
  for (int s = 0; s < nsteps - 1; ++s) {
    v8h na0, na1, nb;
    stage_load((s + 1) << 5, na0, na1, nb);      // overlap with this step's WMMA
    __syncthreads();
    v16h a0 = ld_frag(Ash[cur], tm0, lane);
    v16h a1 = ld_frag(Ash[cur], tm0 + 1, lane);
    v16h b0 = ld_frag(Bsh[cur], tn0, lane);
    v16h b1 = ld_frag(Bsh[cur], tn0 + 1, lane);
    acc[0][0] = wmma_f16(a0, b0, acc[0][0]);
    acc[0][1] = wmma_f16(a0, b1, acc[0][1]);
    acc[1][0] = wmma_f16(a1, b0, acc[1][0]);
    acc[1][1] = wmma_f16(a1, b1, acc[1][1]);
    stage_store(cur ^ 1, na0, na1, nb);
    cur ^= 1;
  }
  __syncthreads();
  {
    v16h a0 = ld_frag(Ash[cur], tm0, lane);
    v16h a1 = ld_frag(Ash[cur], tm0 + 1, lane);
    v16h b0 = ld_frag(Bsh[cur], tn0, lane);
    v16h b1 = ld_frag(Bsh[cur], tn0 + 1, lane);
    acc[0][0] = wmma_f16(a0, b0, acc[0][0]);
    acc[0][1] = wmma_f16(a0, b1, acc[0][1]);
    acc[1][0] = wmma_f16(a1, b0, acc[1][0]);
    acc[1][1] = wmma_f16(a1, b1, acc[1][1]);
  }
  const int nc = lane & 15, g = lane >> 4;
  if (mode == 1) {
#pragma unroll
    for (int mt = 0; mt < 2; ++mt)
#pragma unroll
      for (int j = 0; j < 8; ++j) {
        size_t rb = (size_t)(bm + (tm0 + mt) * 16 + j + 8 * g) * N;
#pragma unroll
        for (int nt = 0; nt < 2; ++nt) {
          int col = bn + (tn0 + nt) * 16 + nc;
          Y[rb + col] = gelu_tanh(acc[mt][nt][j] + bias[col]);
        }
      }
  } else if (mode == 2) {
#pragma unroll
    for (int mt = 0; mt < 2; ++mt)
#pragma unroll
      for (int j = 0; j < 8; ++j) {
        size_t rb = (size_t)(bm + (tm0 + mt) * 16 + j + 8 * g) * N;
#pragma unroll
        for (int nt = 0; nt < 2; ++nt) {
          int col = bn + (tn0 + nt) * 16 + nc;
          Y[rb + col] = acc[mt][nt][j] + bias[col] + res[rb + col];
        }
      }
  } else {
#pragma unroll
    for (int mt = 0; mt < 2; ++mt)
#pragma unroll
      for (int j = 0; j < 8; ++j) {
        size_t rb = (size_t)(bm + (tm0 + mt) * 16 + j + 8 * g) * N;
#pragma unroll
        for (int nt = 0; nt < 2; ++nt) {
          int col = bn + (tn0 + nt) * 16 + nc;
          Y[rb + col] = acc[mt][nt][j] + bias[col];
        }
      }
  }
}

// ---------------------------------------------------------------------------
// LayerNorm over C=512, one block per row.
// ---------------------------------------------------------------------------
__global__ __launch_bounds__(BDIM) void ln_kernel(
    const float* __restrict__ X, const float* __restrict__ gamma,
    const float* __restrict__ beta, float* __restrict__ Y) {
  const size_t row = (size_t)blockIdx.x * 512;
  __shared__ float red[BDIM];
  int tid = threadIdx.x;
  float a0 = X[row + tid], a1 = X[row + tid + 256];
  red[tid] = a0 + a1;
  __syncthreads();
  for (int o = 128; o > 0; o >>= 1) { if (tid < o) red[tid] += red[tid + o]; __syncthreads(); }
  float mean = red[0] * (1.0f / 512.0f);
  __syncthreads();
  float d0 = a0 - mean, d1 = a1 - mean;
  red[tid] = d0 * d0 + d1 * d1;
  __syncthreads();
  for (int o = 128; o > 0; o >>= 1) { if (tid < o) red[tid] += red[tid + o]; __syncthreads(); }
  float rstd = rsqrtf(red[0] * (1.0f / 512.0f) + 1e-5f);
  Y[row + tid]       = d0 * rstd * gamma[tid]       + beta[tid];
  Y[row + tid + 256] = d1 * rstd * gamma[tid + 256] + beta[tid + 256];
}

// ---------------------------------------------------------------------------
// Softmax over contiguous segments of 64 (per-head d=64). Block = 4 segments.
// ---------------------------------------------------------------------------
__global__ __launch_bounds__(BDIM) void softmax64_kernel(float* __restrict__ X) {
  __shared__ float red[4][64];
  const size_t base = (size_t)blockIdx.x * BDIM;
  int tid = threadIdx.x, seg = tid >> 6, e = tid & 63;
  float v = X[base + tid];
  red[seg][e] = v;
  __syncthreads();
  for (int o = 32; o > 0; o >>= 1) { if (e < o) red[seg][e] = fmaxf(red[seg][e], red[seg][e + o]); __syncthreads(); }
  float mx = red[seg][0];
  __syncthreads();
  float ex = __expf(v - mx);
  red[seg][e] = ex;
  __syncthreads();
  for (int o = 32; o > 0; o >>= 1) { if (e < o) red[seg][e] += red[seg][e + o]; __syncthreads(); }
  X[base + tid] = ex / red[seg][0];
}

__global__ __launch_bounds__(BDIM) void zero_kernel(float* __restrict__ p, int n) {
  int i = blockIdx.x * BDIM + threadIdx.x;
  if (i < n) p[i] = 0.0f;
}

// ---------------------------------------------------------------------------
// ctx[bh] += K[b,:,h]^T @ V[b,:,h]  (64x64; T split over blockIdx.y, f32 atomics)
// ksum[bh,d] += sum_t K  (accumulated during staging). Double-buffered LDS.
// Staging thread role: d = tid&63, tb = tid>>6 handles local rows tb*8..tb*8+7.
// ---------------------------------------------------------------------------
__global__ __launch_bounds__(BDIM) void ctx_kernel(
    const float* __restrict__ Kb, const float* __restrict__ Vb,
    float* __restrict__ ctx, float* __restrict__ ksum,
    int T, int C, int rowsPerSplit) {
  const int bh = blockIdx.x;
  const int b = bh >> 3, h = bh & 7;
  const float* Kbase = Kb + (size_t)b * T * C + h * 64;
  const float* Vbase = Vb + (size_t)b * T * C + h * 64;
  const int t0 = blockIdx.y * rowsPerSplit;
  __shared__ alignas(32) _Float16 Ksh[2][4 * 32 * 16];  // A = K^T (d x 32t)
  __shared__ alignas(32) _Float16 Vsh[2][4 * 32 * 16];  // B = V   (32t x d)
  const int tid = threadIdx.x, lane = tid & 31, wave = tid >> 5;
  const int tm = wave >> 1, tn0 = (wave & 1) * 2;
  const int d = tid & 63, tb = tid >> 6;
  const int kdst = ((d >> 4) * 32 + (d & 15) + 16 * (tb & 1)) * 16 + (tb >> 1) * 8;
  const int vdst = ((d >> 4) * 32 + (d & 15) + 16 * (tb & 1)) * 16 + (tb >> 1) * 8;
  v8f acc0 = {}, acc1 = {};
  float ksAcc = 0.0f;

  auto chunk_load = [&](int t, v8h& hk, v8h& hv) {
#pragma unroll
    for (int i = 0; i < 8; ++i) {
      float kf = Kbase[(size_t)(t + tb * 8 + i) * C + d];
      float vf = Vbase[(size_t)(t + tb * 8 + i) * C + d];
      ksAcc += kf;
      hk[i] = (_Float16)kf;
      hv[i] = (_Float16)vf;
    }
  };

  const int nchunks = rowsPerSplit >> 5;
  int cur = 0;
  {
    v8h hk, hv;
    chunk_load(t0, hk, hv);
    *(v8h*)(Ksh[0] + kdst) = hk;
    *(v8h*)(Vsh[0] + vdst) = hv;
  }
  for (int c = 0; c < nchunks - 1; ++c) {
    v8h hk, hv;
    chunk_load(t0 + ((c + 1) << 5), hk, hv);
    __syncthreads();
    v16h a  = ld_frag(Ksh[cur], tm, lane);
    v16h b0 = ld_frag(Vsh[cur], tn0, lane);
    v16h b1 = ld_frag(Vsh[cur], tn0 + 1, lane);
    acc0 = wmma_f16(a, b0, acc0);
    acc1 = wmma_f16(a, b1, acc1);
    *(v8h*)(Ksh[cur ^ 1] + kdst) = hk;
    *(v8h*)(Vsh[cur ^ 1] + vdst) = hv;
    cur ^= 1;
  }
  __syncthreads();
  {
    v16h a  = ld_frag(Ksh[cur], tm, lane);
    v16h b0 = ld_frag(Vsh[cur], tn0, lane);
    v16h b1 = ld_frag(Vsh[cur], tn0 + 1, lane);
    acc0 = wmma_f16(a, b0, acc0);
    acc1 = wmma_f16(a, b1, acc1);
  }
  atomicAdd(&ksum[bh * 64 + d], ksAcc);
  float* cb = ctx + (size_t)bh * 4096;
  const int nc = lane & 15, g = lane >> 4;
#pragma unroll
  for (int j = 0; j < 8; ++j) {
    int m = tm * 16 + j + 8 * g;
    atomicAdd(&cb[m * 64 + tn0 * 16 + nc],       acc0[j]);
    atomicAdd(&cb[m * 64 + (tn0 + 1) * 16 + nc], acc1[j]);
  }
}

// ---------------------------------------------------------------------------
// Out[b,t,h] (init: = q + (q@ctx)*Dinv ; else: += (q@ctx)*Dinv)
// Dinv[t] = 1/(q[t].ksum + eps), computed via staging-time partial dots.
// ---------------------------------------------------------------------------
__global__ __launch_bounds__(BDIM) void attn_out_kernel(
    const float* __restrict__ Q, const float* __restrict__ ctx,
    const float* __restrict__ ksum, float* __restrict__ Out,
    int T, int C, float eps, int init) {
  const int bh = blockIdx.y, b = bh >> 3, h = bh & 7;
  const int t0 = blockIdx.x * 64;
  const float* Qbase = Q + (size_t)b * T * C + h * 64;
  const float* cb = ctx + (size_t)bh * 4096;
  const float* kv = ksum + bh * 64;
  __shared__ alignas(32) _Float16 Qsh[2 * 4 * 32 * 16];  // [kstep][tile][lane][16]
  __shared__ alignas(32) _Float16 Csh[2 * 4 * 32 * 16];
  __shared__ float Dsh[64];
  const int tid = threadIdx.x, lane = tid & 31, wave = tid >> 5;
  if (tid < 64) Dsh[tid] = 0.0f;
  __syncthreads();
#pragma unroll
  for (int i = 0; i < 2; ++i) {
    int asg = tid + i * BDIM;
    int r = asg >> 3, db = asg & 7;
    int ks = db >> 2;
    const float* pq = Qbase + (size_t)(t0 + r) * C + db * 8;
    const float* pk = kv + db * 8;
    v8h hq;
    float s = 0.0f;
#pragma unroll
    for (int e = 0; e < 8; ++e) { float f = pq[e]; s += f * pk[e]; hq[e] = (_Float16)f; }
    int dst = ((ks * 4 + (r >> 4)) * 32 + (r & 15) + 16 * (db & 1)) * 16 + ((db >> 1) & 1) * 8;
    *(v8h*)(Qsh + dst) = hq;
    atomicAdd(&Dsh[r], s);
  }
#pragma unroll
  for (int i = 0; i < 2; ++i) {
    int asg = tid + i * BDIM;
    int dv = asg & 63, kb = asg >> 6;
    int ks = kb >> 2;
    const float* pc = cb + (size_t)(kb * 8) * 64 + dv;
    v8h hc;
#pragma unroll
    for (int e = 0; e < 8; ++e) hc[e] = (_Float16)pc[(size_t)e * 64];
    int dst = ((ks * 4 + (dv >> 4)) * 32 + (dv & 15) + 16 * (kb & 1)) * 16 + ((kb >> 1) & 1) * 8;
    *(v8h*)(Csh + dst) = hc;
  }
  __syncthreads();
  if (tid < 64) Dsh[tid] = 1.0f / (Dsh[tid] + eps);
  __syncthreads();
  const int tm = wave >> 1, tn0 = (wave & 1) * 2;
  v8f acc0 = {}, acc1 = {};
#pragma unroll
  for (int ks = 0; ks < 2; ++ks) {
    const _Float16* qs = Qsh + ks * 2048;
    const _Float16* cs = Csh + ks * 2048;
    v16h a  = ld_frag(qs, tm, lane);
    v16h b0 = ld_frag(cs, tn0, lane);
    v16h b1 = ld_frag(cs, tn0 + 1, lane);
    acc0 = wmma_f16(a, b0, acc0);
    acc1 = wmma_f16(a, b1, acc1);
  }
  const int nc = lane & 15, g = lane >> 4;
  if (init) {
#pragma unroll
    for (int j = 0; j < 8; ++j) {
      int m = tm * 16 + j + 8 * g;
      float dinv = Dsh[m];
      size_t rb = (size_t)(b * T + t0 + m) * C + h * 64;
      size_t qb = (size_t)(t0 + m) * C;
#pragma unroll
      for (int t = 0; t < 2; ++t) {
        int col = (tn0 + t) * 16 + nc;
        Out[rb + col] = (t ? acc1[j] : acc0[j]) * dinv + Qbase[qb + col];
      }
    }
  } else {
#pragma unroll
    for (int j = 0; j < 8; ++j) {
      int m = tm * 16 + j + 8 * g;
      float dinv = Dsh[m];
      size_t rb = (size_t)(b * T + t0 + m) * C + h * 64;
#pragma unroll
      for (int t = 0; t < 2; ++t) {
        int col = (tn0 + t) * 16 + nc;
        Out[rb + col] += (t ? acc1[j] : acc0[j]) * dinv;
      }
    }
  }
}

// ---------------------------------------------------------------------------
extern "C" void kernel_launch(void* const* d_in, const int* in_sizes, int n_in,
                              void* d_out, int out_size, void* d_ws, size_t ws_size,
                              hipStream_t stream) {
  (void)in_sizes; (void)n_in; (void)out_size; (void)ws_size;
  const int B = 4, T = 4096, C = 512, H = 8, D = 64, INNER = 2048;
  const int M = B * T;
  const int SPLITS = 16;

  const float* x     = (const float*)d_in[0];
  const float* ys    = (const float*)d_in[1];
  const float* ln1_g = (const float*)d_in[2];  const float* ln1_b = (const float*)d_in[3];
  const float* ln2_g = (const float*)d_in[4];  const float* ln2_b = (const float*)d_in[5];
  const float* ln3_g = (const float*)d_in[6];  const float* ln3_b = (const float*)d_in[7];
  const float* ln4_g = (const float*)d_in[8];  const float* ln4_b = (const float*)d_in[9];
  const float* ln5_g = (const float*)d_in[10]; const float* ln5_b = (const float*)d_in[11];
  const float* ca_wq = (const float*)d_in[12]; const float* ca_bq = (const float*)d_in[13];
  const float* ca_wk = (const float*)d_in[14]; const float* ca_bk = (const float*)d_in[15];
  const float* ca_wv = (const float*)d_in[16]; const float* ca_bv = (const float*)d_in[17];
  const float* ca_wo = (const float*)d_in[18]; const float* ca_bo = (const float*)d_in[19];
  const float* sa_wq = (const float*)d_in[20]; const float* sa_bq = (const float*)d_in[21];
  const float* sa_wk = (const float*)d_in[22]; const float* sa_bk = (const float*)d_in[23];
  const float* sa_wv = (const float*)d_in[24]; const float* sa_bv = (const float*)d_in[25];
  const float* sa_wo = (const float*)d_in[26]; const float* sa_bo = (const float*)d_in[27];
  const float* f1w1  = (const float*)d_in[28]; const float* f1b1  = (const float*)d_in[29];
  const float* f1w2  = (const float*)d_in[30]; const float* f1b2  = (const float*)d_in[31];
  const float* f2w1  = (const float*)d_in[32]; const float* f2b1  = (const float*)d_in[33];
  const float* f2w2  = (const float*)d_in[34]; const float* f2b2  = (const float*)d_in[35];
  float* out = (float*)d_out;

  const size_t MC = (size_t)M * C;
  float* ws   = (float*)d_ws;
  float* xn   = ws;
  float* q    = xn + MC;
  float* kb   = q + MC;
  float* vb   = kb + MC;
  float* attn = vb + MC;
  float* xa   = attn + MC;
  float* xb2  = xa + MC;
  float* hbuf = xb2 + MC;
  float* ksum = hbuf + (size_t)M * INNER;
  float* ctx  = ksum + B * H * D;

  dim3 blk(BDIM);
  dim3 gLN(M);
  dim3 gSm((unsigned)(MC / BDIM));
  dim3 gGemmC(M / 128, C / 64);
  dim3 gGemmI(M / 128, INNER / 64);
  dim3 gCtx(B * H, SPLITS);
  dim3 gAo(T / 64, B * H);
  const int zn = B * H * D + B * H * D * D;
  dim3 gZero((zn + BDIM - 1) / BDIM);

  // ---- linear cross attention ----
  ln_kernel<<<gLN, blk, 0, stream>>>(x, ln1_g, ln1_b, xn);
  gemm_kernel<<<gGemmC, blk, 0, stream>>>(xn, ca_wq, ca_bq, nullptr, q, M, C, C, 0);
  softmax64_kernel<<<gSm, blk, 0, stream>>>(q);
  for (int i = 0; i < 2; ++i) {
    const float* yi = ys + (size_t)i * MC;
    ln_kernel<<<gLN, blk, 0, stream>>>(yi, ln2_g + i * C, ln2_b + i * C, xn);
    gemm_kernel<<<gGemmC, blk, 0, stream>>>(xn, ca_wk + (size_t)i * C * C, ca_bk + i * C,
                                            nullptr, kb, M, C, C, 0);
    softmax64_kernel<<<gSm, blk, 0, stream>>>(kb);
    gemm_kernel<<<gGemmC, blk, 0, stream>>>(xn, ca_wv + (size_t)i * C * C, ca_bv + i * C,
                                            nullptr, vb, M, C, C, 0);
    zero_kernel<<<gZero, blk, 0, stream>>>(ksum, zn);
    ctx_kernel<<<gCtx, blk, 0, stream>>>(kb, vb, ctx, ksum, T, C, T / SPLITS);
    attn_out_kernel<<<gAo, blk, 0, stream>>>(q, ctx, ksum, attn, T, C, 1e-8f, i == 0);
  }
  gemm_kernel<<<gGemmC, blk, 0, stream>>>(attn, ca_wo, ca_bo, x, xa, M, C, C, 2);

  // ---- FFN 1 ----
  ln_kernel<<<gLN, blk, 0, stream>>>(xa, ln3_g, ln3_b, xn);
  gemm_kernel<<<gGemmI, blk, 0, stream>>>(xn, f1w1, f1b1, nullptr, hbuf, M, INNER, C, 1);
  gemm_kernel<<<gGemmC, blk, 0, stream>>>(hbuf, f1w2, f1b2, xa, xb2, M, C, INNER, 2);

  // ---- linear self attention ('l1': eps = 0) ----
  ln_kernel<<<gLN, blk, 0, stream>>>(xb2, ln4_g, ln4_b, xn);
  gemm_kernel<<<gGemmC, blk, 0, stream>>>(xn, sa_wq, sa_bq, nullptr, q, M, C, C, 0);
  softmax64_kernel<<<gSm, blk, 0, stream>>>(q);
  gemm_kernel<<<gGemmC, blk, 0, stream>>>(xn, sa_wk, sa_bk, nullptr, kb, M, C, C, 0);
  softmax64_kernel<<<gSm, blk, 0, stream>>>(kb);
  gemm_kernel<<<gGemmC, blk, 0, stream>>>(xn, sa_wv, sa_bv, nullptr, vb, M, C, C, 0);
  zero_kernel<<<gZero, blk, 0, stream>>>(ksum, zn);
  ctx_kernel<<<gCtx, blk, 0, stream>>>(kb, vb, ctx, ksum, T, C, T / SPLITS);
  attn_out_kernel<<<gAo, blk, 0, stream>>>(q, ctx, ksum, attn, T, C, 0.0f, 1);
  gemm_kernel<<<gGemmC, blk, 0, stream>>>(attn, sa_wo, sa_bo, xb2, xa, M, C, C, 2);

  // ---- FFN 2 ----
  ln_kernel<<<gLN, blk, 0, stream>>>(xa, ln5_g, ln5_b, xn);
  gemm_kernel<<<gGemmI, blk, 0, stream>>>(xn, f2w1, f2b1, nullptr, hbuf, M, INNER, C, 1);
  gemm_kernel<<<gGemmC, blk, 0, stream>>>(hbuf, f2w2, f2b2, xa, out, M, C, INNER, 2);
}
